// SlidingWindowAttention_71622874628686
// MI455X (gfx1250) — matrix operand order repaired and checked by reading, not compiled, not run on hardware
//
#include <hip/hip_runtime.h>
#include <math.h>

typedef __bf16 bf16;
typedef __bf16 v16bf __attribute__((ext_vector_type(16)));
typedef __bf16 v8bf  __attribute__((ext_vector_type(8)));
typedef __bf16 v4bf  __attribute__((ext_vector_type(4)));
typedef float  v8f   __attribute__((ext_vector_type(8)));
typedef float  v4f   __attribute__((ext_vector_type(4)));

constexpr int T    = 4096;
constexpr int D    = 1024;
constexpr int H    = 16;
constexpr int HD   = 64;
constexpr int WIN  = 256;
constexpr int NQKV = 3 * D;

// ---------- WMMA helpers ----------
// A fragment (16x32 bf16, MxK): lane l holds row m=l%16.
// elements 0..15 -> k = 2*base + e      (base = 8*(l/16))
__device__ __forceinline__ v16bf load_a_frag(const bf16* rowptr, int base) {
  v16bf a;
  *(v8bf*)&a       = *(const v8bf*)(rowptr + 2 * base);
  *((v8bf*)&a + 1) = *(const v8bf*)(rowptr + 2 * base + 8);
  return a;
}

// B fragment (32x16 bf16, KxN): lane l holds column n=l%16,
// elements e=0..15 -> k = 16*(l/16) + e  (caller bakes the 16*hi offset in).
__device__ __forceinline__ v16bf load_b_frag(const bf16* p) {
  v16bf b;
  *(v8bf*)&b       = *(const v8bf*)(p);
  *((v8bf*)&b + 1) = *(const v8bf*)(p + 8);
  return b;
}

__device__ __forceinline__ v8f wmma_bf16(v16bf a, v16bf b, v8f c) {
  return __builtin_amdgcn_wmma_f32_16x16x32_bf16(false, a, false, b,
                                                 (short)0, c, false, false);
}

// ---------- fp32 -> bf16 conversion (4 elems / thread) ----------
__global__ void to_bf16_kernel(const float* __restrict__ in, bf16* __restrict__ out, int n4) {
  int i = blockIdx.x * blockDim.x + threadIdx.x;
  if (i < n4) {
    v4f f = *(const v4f*)(in + 4 * (size_t)i);
    v4bf b;
    b[0] = (bf16)f[0]; b[1] = (bf16)f[1]; b[2] = (bf16)f[2]; b[3] = (bf16)f[3];
    *(v4bf*)(out + 4 * (size_t)i) = b;
  }
}

// ---------- 32x64 per-wave GEMM core, double-buffered K loop ----------
// acc[g][t]: rows m0+16g .. +15, cols n0+16t .. +15.  K must be multiple of 64.
__device__ __forceinline__ void gemm_32x64(const bf16* __restrict__ A,
                                           const bf16* __restrict__ Bt,
                                           int m0, int n0, int K,
                                           int n, int hi, v8f acc[2][4]) {
  const int base = 8 * hi;
  const bf16* arow0 = A + (size_t)(m0 + n) * K;
  const bf16* arow1 = A + (size_t)(m0 + 16 + n) * K;
  const bf16* brow[4];
#pragma unroll
  for (int t = 0; t < 4; ++t)
    brow[t] = Bt + (size_t)(n0 + t * 16 + n) * K + 16 * hi;

  // stage-even fragments (kc = 0)
  v16bf a0 = load_a_frag(arow0, base);
  v16bf a1 = load_a_frag(arow1, base);
  v16bf w[4];
#pragma unroll
  for (int t = 0; t < 4; ++t) w[t] = load_b_frag(brow[t]);

  for (int kc = 0; kc < K; kc += 64) {
    // prefetch stage-odd (kc+32) while computing stage-even
    v16bf a0o = load_a_frag(arow0 + kc + 32, base);
    v16bf a1o = load_a_frag(arow1 + kc + 32, base);
    v16bf u[4];
#pragma unroll
    for (int t = 0; t < 4; ++t) u[t] = load_b_frag(brow[t] + kc + 32);

#pragma unroll
    for (int t = 0; t < 4; ++t) {
      acc[0][t] = wmma_bf16(a0, w[t], acc[0][t]);
      acc[1][t] = wmma_bf16(a1, w[t], acc[1][t]);
    }

    const int kn = kc + 64;                 // prefetch next stage-even
    if (kn < K) {
      a0 = load_a_frag(arow0 + kn, base);
      a1 = load_a_frag(arow1 + kn, base);
#pragma unroll
      for (int t = 0; t < 4; ++t) w[t] = load_b_frag(brow[t] + kn);
    }

#pragma unroll
    for (int t = 0; t < 4; ++t) {
      acc[0][t] = wmma_bf16(a0o, u[t], acc[0][t]);
      acc[1][t] = wmma_bf16(a1o, u[t], acc[1][t]);
    }
  }
}

// ---------- QKV GEMM: scatter epilogue into Q (scaled), K, and V^T ----------
__global__ __launch_bounds__(32) void qkv_gemm(const bf16* __restrict__ X,
                                               const bf16* __restrict__ W,
                                               bf16* __restrict__ Qb,
                                               bf16* __restrict__ Kb,
                                               bf16* __restrict__ Vt) {
  const int lane = threadIdx.x;
  const int n    = lane & 15;
  const int hi   = lane >> 4;
  const int m0   = blockIdx.x * 32;
  const int n0   = blockIdx.y * 64;

  v8f acc[2][4] = {};
  gemm_32x64(X, W, m0, n0, D, n, hi, acc);

#pragma unroll
  for (int t = 0; t < 4; ++t) {
    const int c     = n0 + t * 16 + n;
    const int which = c / D;           // 0=q 1=k 2=v
    const int cc    = c % D;
    const int h     = cc / HD;
    const int d     = cc % HD;
#pragma unroll
    for (int g = 0; g < 2; ++g)
#pragma unroll
      for (int r = 0; r < 8; ++r) {
        const int row = m0 + 16 * g + r + 8 * hi;
        const float v = acc[g][t][r];
        if (which == 0)      Qb[((size_t)h * T + row) * HD + d] = (bf16)(v * 0.125f);
        else if (which == 1) Kb[((size_t)h * T + row) * HD + d] = (bf16)v;
        else                 Vt[((size_t)h * HD + d) * T + row] = (bf16)v;
      }
  }
}

// ---------- Sliding-window flash attention ----------
__global__ __launch_bounds__(32) void attn_kernel(const bf16* __restrict__ Qb,
                                                  const bf16* __restrict__ Kb,
                                                  const bf16* __restrict__ Vt,
                                                  bf16* __restrict__ attn) {
  __shared__ bf16 plds[16 * 32];   // P tile, row-major [16 q][32 keys]
  const int lane = threadIdx.x;
  const int n    = lane & 15;
  const int hi   = lane >> 4;
  const int base = 8 * hi;
  const int h    = blockIdx.x >> 8;          // / 256
  const int q0   = (blockIdx.x & 255) * 16;

  const bf16* qrow = Qb + ((size_t)h * T + q0 + n) * HD;
  v16bf qa0 = load_a_frag(qrow, base);
  v16bf qa1 = load_a_frag(qrow + 32, base);

  const bf16* kbase = Kb + (size_t)h * T * HD + (size_t)n * HD + 16 * hi;
  const bf16* vbase = Vt + (size_t)h * HD * T + (size_t)n * T + 16 * hi;

  float mrun[8], lrun[8];
#pragma unroll
  for (int r = 0; r < 8; ++r) { mrun[r] = -INFINITY; lrun[r] = 0.0f; }
  v8f o[4] = {};

  int jlo = q0 - (WIN - 1); if (jlo < 0) jlo = 0;
  const int jt0 = (jlo / 32) * 32;

  for (int jt = jt0; jt <= q0 + 15; jt += 32) {
    // ---- K tiles: keys jt..jt+15 (left) and jt+16..jt+31 (right) ----
    const bf16* klrow = kbase + (size_t)jt * HD;
    const bf16* krrow = klrow + (size_t)16 * HD;
    v8f sl = {}, sr = {};
    sl = wmma_bf16(qa0, load_b_frag(klrow),      sl);
    sl = wmma_bf16(qa1, load_b_frag(klrow + 32), sl);
    sr = wmma_bf16(qa0, load_b_frag(krrow),      sr);
    sr = wmma_bf16(qa1, load_b_frag(krrow + 32), sr);

    // ---- V fragments hoisted: latency hides behind softmax ----
    v16bf vb[4];
#pragma unroll
    for (int t = 0; t < 4; ++t)
      vb[t] = load_b_frag(vbase + (size_t)t * 16 * T + jt);

    // prefetch next chunk's K rows (predictable stride)
    if (jt + 32 <= q0 + 15) {
      __builtin_prefetch(klrow + (size_t)32 * HD, 0, 3);
      __builtin_prefetch(krrow + (size_t)32 * HD, 0, 3);
    }

    // ---- mask + online softmax (row m lives on 16 lanes of one half) ----
    float alpha[8];
    const int jl = jt + n, jr = jt + 16 + n;
#pragma unroll
    for (int r = 0; r < 8; ++r) {
      const int i = q0 + r + 8 * hi;
      float svl = ((jl <= i) && (jl >= i - (WIN - 1))) ? sl[r] : -1.0e30f;
      float svr = ((jr <= i) && (jr >= i - (WIN - 1))) ? sr[r] : -1.0e30f;
      float mx = fmaxf(svl, svr);
      mx = fmaxf(mx, __shfl_xor(mx, 1, 32));
      mx = fmaxf(mx, __shfl_xor(mx, 2, 32));
      mx = fmaxf(mx, __shfl_xor(mx, 4, 32));
      mx = fmaxf(mx, __shfl_xor(mx, 8, 32));
      const float mn = fmaxf(mrun[r], mx);
      const float sc = __expf(mrun[r] - mn);
      const float pl = __expf(svl - mn);
      const float pr = __expf(svr - mn);
      float rs = pl + pr;
      rs += __shfl_xor(rs, 1, 32);
      rs += __shfl_xor(rs, 2, 32);
      rs += __shfl_xor(rs, 4, 32);
      rs += __shfl_xor(rs, 8, 32);
      lrun[r] = lrun[r] * sc + rs;
      mrun[r] = mn;
      alpha[r] = sc;
      plds[(r + 8 * hi) * 32 + n]      = (bf16)pl;
      plds[(r + 8 * hi) * 32 + 16 + n] = (bf16)pr;
    }
#pragma unroll
    for (int t = 0; t < 4; ++t)
#pragma unroll
      for (int r = 0; r < 8; ++r) o[t][r] *= alpha[r];

    __syncthreads();                       // order LDS P stores vs loads
    v16bf pa = load_a_frag(plds + n * 32, base);   // A-layout P (16x32)
#pragma unroll
    for (int t = 0; t < 4; ++t) o[t] = wmma_bf16(pa, vb[t], o[t]);
    __syncthreads();                       // protect plds before next chunk
  }

#pragma unroll
  for (int r = 0; r < 8; ++r) {
    const float inv = 1.0f / lrun[r];
    const int row = q0 + r + 8 * hi;
#pragma unroll
    for (int t = 0; t < 4; ++t)
      attn[(size_t)row * D + h * HD + t * 16 + n] = (bf16)(o[t][r] * inv);
  }
}

// ---------- Output projection ----------
__global__ __launch_bounds__(32) void proj_gemm(const bf16* __restrict__ A,
                                                const bf16* __restrict__ W,
                                                float* __restrict__ out) {
  const int lane = threadIdx.x;
  const int n    = lane & 15;
  const int hi   = lane >> 4;
  const int m0   = blockIdx.x * 32;
  const int n0   = blockIdx.y * 64;

  v8f acc[2][4] = {};
  gemm_32x64(A, W, m0, n0, D, n, hi, acc);

#pragma unroll
  for (int t = 0; t < 4; ++t)
#pragma unroll
    for (int g = 0; g < 2; ++g)
#pragma unroll
      for (int r = 0; r < 8; ++r)
        out[(size_t)(m0 + 16 * g + r + 8 * hi) * D + n0 + t * 16 + n] = acc[g][t][r];
}

extern "C" void kernel_launch(void* const* d_in, const int* in_sizes, int n_in,
                              void* d_out, int out_size, void* d_ws, size_t ws_size,
                              hipStream_t stream) {
  (void)in_sizes; (void)n_in; (void)out_size; (void)ws_size;
  const float* x     = (const float*)d_in[0];
  const float* wqkv  = (const float*)d_in[1];
  const float* wproj = (const float*)d_in[2];
  float* out = (float*)d_out;

  char* ws = (char*)d_ws;
  bf16* xb    = (bf16*)ws; ws += (size_t)T * D * 2;
  bf16* wqb   = (bf16*)ws; ws += (size_t)NQKV * D * 2;
  bf16* wpb   = (bf16*)ws; ws += (size_t)D * D * 2;
  bf16* Qb    = (bf16*)ws; ws += (size_t)H * T * HD * 2;
  bf16* Kb    = (bf16*)ws; ws += (size_t)H * T * HD * 2;
  bf16* Vtb   = (bf16*)ws; ws += (size_t)H * HD * T * 2;
  bf16* attnb = (bf16*)ws; ws += (size_t)T * D * 2;

  const int nx4 = T * D / 4, nq4 = NQKV * D / 4, np4 = D * D / 4;
  to_bf16_kernel<<<(nx4 + 255) / 256, 256, 0, stream>>>(x, xb, nx4);
  to_bf16_kernel<<<(nq4 + 255) / 256, 256, 0, stream>>>(wqkv, wqb, nq4);
  to_bf16_kernel<<<(np4 + 255) / 256, 256, 0, stream>>>(wproj, wpb, np4);

  qkv_gemm<<<dim3(T / 32, NQKV / 64), 32, 0, stream>>>(xb, wqb, Qb, Kb, Vtb);
  attn_kernel<<<dim3(H * (T / 16)), 32, 0, stream>>>(Qb, Kb, Vtb, attnb);
  proj_gemm<<<dim3(T / 32, D / 64), 32, 0, stream>>>(attnb, wpb, out);
}
